// Quantize_36876589203895
// MI455X (gfx1250) — compile-verified, matrix operand order checked by reading, not run on hardware
//
#include <hip/hip_runtime.h>

// ---------------- types ----------------
typedef __attribute__((ext_vector_type(8)))  float  v8f;
typedef __attribute__((ext_vector_type(16))) __bf16 v16bf;

// ---------------- problem constants ----------------
#define NROW   65536       // 64*32*32 flattened rows
#define DIM    256
#define NE     2048
#define DECAYF 0.99f
#define EPSF   1e-5f

// ---------------- output layout (floats, concatenated in return order) ----
#define OFF_Q    0                       // quantize_st: 16777216
#define OFF_DIFF 16777216                // diff: 1
#define OFF_IND  16777217                // embed_ind: 65536
#define OFF_NEMB 16842753                // new_embed: 524288
#define OFF_NCS  17367041                // new_cluster_size: 2048
#define OFF_NEA  17369089                // new_embed_avg: 524288

// ---------------- workspace layout (bytes) ----------------
#define WS_IND   0                       // int[65536]
#define WS_CNT   262144                  // int[2048]
#define WS_ESUM  270336                  // float[524288]
#define WS_DIFFP 2367488                 // float[65536] per-row diff partials
#define WS_N     2629632                 // float[1]
#define WS_SQN   2629888                 // float[2048]
#define WS_BHI   2638080                 // ushort[524288] bf16-hi of embed, B-packed
#define WS_BLO   3686656                 // ushort[524288] bf16-lo of embed, B-packed

// bf16 split helpers (round-to-nearest-even)
__device__ __forceinline__ unsigned short bf_hi(float f) {
    unsigned int u = __float_as_uint(f);
    u += 0x7fffu + ((u >> 16) & 1u);
    return (unsigned short)(u >> 16);
}
__device__ __forceinline__ float bf_val(unsigned short s) {
    return __uint_as_float(((unsigned int)s) << 16);
}

// ---- CDNA5 async copy (global -> LDS), tracked with ASYNCcnt ----
__device__ __forceinline__ void async_b128(unsigned lds_byte, const void* g) {
    asm volatile("global_load_async_to_lds_b128 %0, %1, off"
                 :: "v"(lds_byte), "v"(g) : "memory");
}
__device__ __forceinline__ void wait_async0() {
    asm volatile("s_wait_asynccnt 0" ::: "memory");
}
__device__ __forceinline__ void wait_ds0() {
    asm volatile("s_wait_dscnt 0" ::: "memory");
}

// ============================================================
// Prep: per-code squared norms  sqn[j] = sum_d embed[d][j]^2
// ============================================================
__global__ __launch_bounds__(256) void vq_prep_sqn(const float* __restrict__ embed,
                                                   float* __restrict__ sqn) {
    int j = blockIdx.x * 256 + threadIdx.x;     // 0..2047
    float s = 0.f;
    for (int k = 0; k < DIM; ++k) {
        float e = embed[k * NE + j];
        s += e * e;
    }
    sqn[j] = s;
}

// ============================================================
// Prep: repack embed into WMMA B-matrix lane/K layout, bf16 hi/lo.
// Layout: bpack[cb(128)][kb(8)][lane(32)][e(16)]
//   lane half h = lane>>4, col = cb*16 + (lane&15)
//   K = kb*32 + (e<8 ? 8h+e : 16+8h+(e-8))   (mirrors ISA 16-bit A layout)
// ============================================================
__global__ __launch_bounds__(256) void vq_prep_bpack(const float* __restrict__ embed,
                                                     unsigned short* __restrict__ bhi,
                                                     unsigned short* __restrict__ blo) {
    int p = blockIdx.x * 256 + threadIdx.x;     // 0..524287
    int e    = p & 15;
    int lane = (p >> 4) & 31;
    int kb   = (p >> 9) & 7;
    int cb   = p >> 12;
    int h = lane >> 4;
    int m = lane & 15;
    int kk = (e < 8) ? (8 * h + e) : (16 + 8 * h + (e - 8));
    int K = kb * 32 + kk;
    int col = cb * 16 + m;
    float f = embed[K * NE + col];
    unsigned short hi = bf_hi(f);
    bhi[p] = hi;
    blo[p] = bf_hi(f - bf_val(hi));
}

// ============================================================
// Main: fused GEMM(score) + argmin.
// 256 threads = 8 waves; wave w owns rows [rowBase+16w, rowBase+16w+16).
// A fragments (x rows, bf16 hi/lo) held in 128 VGPRs for the whole sweep;
// B (codes) double-buffered in LDS via global_load_async_to_lds_b128,
// prefetching block sb+1 while block sb's 24 WMMAs execute.
// score = ||embed_j||^2 - 2*dot  (row-norm dropped: constant per row)
// ============================================================
__global__ __launch_bounds__(256) void vq_argmin_gemm(const float* __restrict__ x,
                                                      const unsigned short* __restrict__ bhi,
                                                      const unsigned short* __restrict__ blo,
                                                      const float* __restrict__ sqn,
                                                      int* __restrict__ indOut,
                                                      float* __restrict__ out) {
    // 128 KB, two lives:
    //   phase 1: A-pack  hi -> [0..32767], lo -> [32768..65535]   (ushort idx)
    //   phase 2: B double buffer: buf b at ushort [b*8192 .. b*8192+8191]
    //            (hi half [0..4095], lo half [4096..8191] within a buffer)
    __shared__ __align__(32) unsigned short smem[65536];

    const int tid  = threadIdx.x;
    const int lane = tid & 31;
    const int w    = tid >> 5;
    const int rowBase = blockIdx.x * 128;

    // ---- Phase 1: load 128 rows of x, split to bf16 hi/lo, pack in A layout
    const float4* xv = (const float4*)(x + (size_t)rowBase * DIM);
    for (int it = 0; it < 32; ++it) {
        int i = it * 256 + tid;                 // float4 index 0..8191
        float4 v = xv[i];
        int fbase = i * 4;
        int row = fbase >> 8;                   // 0..127
        int k0  = fbase & 255;
        float vv[4] = {v.x, v.y, v.z, v.w};
        #pragma unroll
        for (int q = 0; q < 4; ++q) {
            int K  = k0 + q;
            int kb = K >> 5, kk = K & 31;
            int h  = (kk >> 3) & 1;
            int e  = (kk & 7) + ((kk >> 4) << 3);
            int ln = (row & 15) + (h << 4);
            int idx = (((row >> 4) * 8 + kb) * 32 + ln) * 16 + e;
            unsigned short hi = bf_hi(vv[q]);
            smem[idx]         = hi;
            smem[32768 + idx] = bf_hi(vv[q] - bf_val(hi));
        }
    }
    __syncthreads();

    // ---- Hoist this wave's A fragments into registers (16 x v16bf = 128 VGPRs)
    v16bf aH[8], aL[8];
    #pragma unroll
    for (int kb = 0; kb < 8; ++kb) {
        int idx = ((w * 8 + kb) * 32 + lane) * 16;
        aH[kb] = *(const v16bf*)&smem[idx];
        aL[kb] = *(const v16bf*)&smem[32768 + idx];
    }
    wait_ds0();          // fragment loads landed in VGPRs
    __syncthreads();     // all waves done with the A region before B DMA reuses it

    const unsigned ldsBase = (unsigned)(size_t)(const void*)&smem[0];

    // ---- prologue: async-stage code block 0 into buffer 0
    {
        const char* gh = (const char*)bhi;      // 8 KB hi
        const char* gl = (const char*)blo;      // 8 KB lo
        async_b128(ldsBase + tid * 16,                gh + tid * 16);
        async_b128(ldsBase + (tid + 256) * 16,        gh + (tid + 256) * 16);
        async_b128(ldsBase + 8192 + tid * 16,         gl + tid * 16);
        async_b128(ldsBase + 8192 + (tid + 256) * 16, gl + (tid + 256) * 16);
    }
    wait_async0();
    __syncthreads();

    float runMin[8];
    int   runIdx[8];
    #pragma unroll
    for (int e = 0; e < 8; ++e) { runMin[e] = 3.0e38f; runIdx[e] = 0; }

    // ---- Phase 2: sweep all 2048 codes, 16 at a time, double-buffered
    for (int sb = 0; sb < 128; ++sb) {
        int buf = sb & 1;

        // prefetch next code block into the other buffer (overlaps the WMMAs)
        if (sb + 1 < 128) {
            unsigned dst = ldsBase + (buf ^ 1) * 16384;
            const char* gh = (const char*)bhi + (size_t)(sb + 1) * 8192;
            const char* gl = (const char*)blo + (size_t)(sb + 1) * 8192;
            async_b128(dst + tid * 16,                gh + tid * 16);
            async_b128(dst + (tid + 256) * 16,        gh + (tid + 256) * 16);
            async_b128(dst + 8192 + tid * 16,         gl + tid * 16);
            async_b128(dst + 8192 + (tid + 256) * 16, gl + (tid + 256) * 16);
        }

        const int base = buf * 8192;            // ushort units
        v8f acc = {0.f, 0.f, 0.f, 0.f, 0.f, 0.f, 0.f, 0.f};
        #pragma unroll
        for (int kb = 0; kb < 8; ++kb) {
            v16bf bh = *(const v16bf*)&smem[base + (kb * 32 + lane) * 16];
            v16bf bl = *(const v16bf*)&smem[base + 4096 + (kb * 32 + lane) * 16];
            acc = __builtin_amdgcn_wmma_f32_16x16x32_bf16(false, aH[kb], false, bh,
                                                          (short)0, acc, false, false);
            acc = __builtin_amdgcn_wmma_f32_16x16x32_bf16(false, aH[kb], false, bl,
                                                          (short)0, acc, false, false);
            acc = __builtin_amdgcn_wmma_f32_16x16x32_bf16(false, aL[kb], false, bh,
                                                          (short)0, acc, false, false);
        }

        float sq = sqn[sb * 16 + (lane & 15)];  // hot in cache, per-16-col block
        int col  = sb * 16 + (lane & 15);
        #pragma unroll
        for (int e = 0; e < 8; ++e) {
            float s = sq - 2.0f * acc[e];
            if (s < runMin[e]) { runMin[e] = s; runIdx[e] = col; }
        }

        wait_ds0();       // my LDS reads of buf are complete
        wait_async0();    // my prefetch of buf^1 is complete
        __syncthreads();  // everyone, before buf is overwritten next iteration
    }

    // ---- Phase 3: argmin across the 16 lanes of each half (ties -> lowest idx)
    #pragma unroll
    for (int off = 1; off < 16; off <<= 1) {
        #pragma unroll
        for (int e = 0; e < 8; ++e) {
            float om = __shfl_xor(runMin[e], off, 32);
            int   oi = __shfl_xor(runIdx[e], off, 32);
            if (om < runMin[e] || (om == runMin[e] && oi < runIdx[e])) {
                runMin[e] = om; runIdx[e] = oi;
            }
        }
    }
    if ((lane & 15) == 0) {
        // C layout: VGPR e is row e (lanes 0-15) / row 8+e (lanes 16-31)
        int rbase = rowBase + w * 16 + ((lane >> 4) << 3);
        #pragma unroll
        for (int e = 0; e < 8; ++e) {
            indOut[rbase + e] = runIdx[e];
            out[OFF_IND + rbase + e] = (float)runIdx[e];
        }
    }
}

// ============================================================
// Quantize + straight-through + diff partials + histogram + segment-sum.
// One block per row (65536 blocks x 256 threads, thread = dim).
// ============================================================
__global__ __launch_bounds__(256) void vq_scatter(const float* __restrict__ x,
                                                  const float* __restrict__ embed,
                                                  const int* __restrict__ ind,
                                                  float* __restrict__ out,
                                                  float* __restrict__ esum,
                                                  int* __restrict__ cnt,
                                                  float* __restrict__ diffp) {
    __shared__ float red[256];
    int row = blockIdx.x;
    int d   = threadIdx.x;
    int j   = ind[row];
    size_t gid = (size_t)row * DIM + d;
    float xvv = x[gid];
    float q  = embed[d * NE + j];            // embed.T row j == embed column j
    out[OFF_Q + gid] = xvv + (q - xvv);      // straight-through == quantize
    float dq = q - xvv;
    red[d] = dq * dq;
    atomicAdd(&esum[d * NE + j], xvv);       // embed_sum[d][j] += x[row][d]
    if (d == 0) atomicAdd(&cnt[j], 1);
    __syncthreads();
    for (int s = 128; s > 0; s >>= 1) {
        if (d < s) red[d] += red[d + s];
        __syncthreads();
    }
    if (d == 0) diffp[row] = red[0];
}

// ============================================================
// Small finalize (single block, deterministic reductions):
//   diff = sum(diffp)/N ; new_cluster_size ; n = sum(new_cluster_size)
// ============================================================
__global__ __launch_bounds__(256) void vq_fin_small(const float* __restrict__ cluster_size,
                                                    const int* __restrict__ cnt,
                                                    const float* __restrict__ diffp,
                                                    float* __restrict__ out,
                                                    float* __restrict__ wn) {
    __shared__ float red[256];
    int t = threadIdx.x;

    float s = 0.f;
    for (int i = t; i < NROW; i += 256) s += diffp[i];
    red[t] = s;
    __syncthreads();
    for (int st = 128; st > 0; st >>= 1) {
        if (t < st) red[t] += red[t + st];
        __syncthreads();
    }
    if (t == 0) out[OFF_DIFF] = red[0] / ((float)NROW * (float)DIM);
    __syncthreads();

    float ln = 0.f;
    for (int jj = t; jj < NE; jj += 256) {
        float v = cluster_size[jj] * DECAYF + (1.0f - DECAYF) * (float)cnt[jj];
        out[OFF_NCS + jj] = v;
        ln += v;
    }
    red[t] = ln;
    __syncthreads();
    for (int st = 128; st > 0; st >>= 1) {
        if (t < st) red[t] += red[t + st];
        __syncthreads();
    }
    if (t == 0) wn[0] = red[0];
}

// ============================================================
// Elementwise finalize over [256 x 2048]: new_embed_avg, new_embed.
// ============================================================
__global__ __launch_bounds__(256) void vq_fin_embed(const float* __restrict__ embed_avg,
                                                    const float* __restrict__ esum,
                                                    const float* __restrict__ out_ncs,
                                                    const float* __restrict__ wn,
                                                    float* __restrict__ out) {
    int gid = blockIdx.x * 256 + threadIdx.x;   // 0..524287
    int j = gid & (NE - 1);
    float nea = embed_avg[gid] * DECAYF + (1.0f - DECAYF) * esum[gid];
    out[OFF_NEA + gid] = nea;
    float ncs = out_ncs[j];
    float n = wn[0];
    float cs = (ncs + EPSF) / (n + (float)NE * EPSF) * n;
    out[OFF_NEMB + gid] = nea / cs;
}

// ============================================================
extern "C" void kernel_launch(void* const* d_in, const int* in_sizes, int n_in,
                              void* d_out, int out_size, void* d_ws, size_t ws_size,
                              hipStream_t stream) {
    const float* x            = (const float*)d_in[0];
    const float* embed        = (const float*)d_in[1];
    const float* cluster_size = (const float*)d_in[2];
    const float* embed_avg    = (const float*)d_in[3];
    float* out = (float*)d_out;

    char* ws = (char*)d_ws;
    int*            ws_ind   = (int*)           (ws + WS_IND);
    int*            ws_cnt   = (int*)           (ws + WS_CNT);
    float*          ws_esum  = (float*)         (ws + WS_ESUM);
    float*          ws_diffp = (float*)         (ws + WS_DIFFP);
    float*          ws_n     = (float*)         (ws + WS_N);
    float*          ws_sqn   = (float*)         (ws + WS_SQN);
    unsigned short* ws_bhi   = (unsigned short*)(ws + WS_BHI);
    unsigned short* ws_blo   = (unsigned short*)(ws + WS_BLO);

    // zero histogram + segment-sum accumulators (contiguous region)
    hipMemsetAsync(ws + WS_CNT, 0, (WS_DIFFP - WS_CNT), stream);

    vq_prep_sqn  <<<NE / 256, 256, 0, stream>>>(embed, ws_sqn);
    vq_prep_bpack<<<(NE * DIM) / 256, 256, 0, stream>>>(embed, ws_bhi, ws_blo);
    vq_argmin_gemm<<<NROW / 128, 256, 0, stream>>>(x, ws_bhi, ws_blo, ws_sqn,
                                                   ws_ind, out);
    vq_scatter   <<<NROW, 256, 0, stream>>>(x, embed, ws_ind, out,
                                            ws_esum, ws_cnt, ws_diffp);
    vq_fin_small <<<1, 256, 0, stream>>>(cluster_size, ws_cnt, ws_diffp, out, ws_n);
    vq_fin_embed <<<(DIM * NE) / 256, 256, 0, stream>>>(embed_avg, ws_esum,
                                                        out + OFF_NCS, ws_n, out);
}
